// GraphBasedRelationModel_32856499814708
// MI455X (gfx1250) — matrix operand order repaired and checked by reading, not compiled, not run on hardware
//
#include <hip/hip_runtime.h>

// ---------------------------------------------------------------------------
// CDNA5 (gfx1250) WMMA bf16 implementation of the 3-layer graph relation model.
// D = A*B + C via v_wmma_f32_16x16x32_bf16, wave32, f32 accumulation.
// Double-buffered LDS pipeline; async global->LDS copies (ASYNCcnt).
// ---------------------------------------------------------------------------

typedef __attribute__((ext_vector_type(16))) __bf16 v16bf;
typedef __attribute__((ext_vector_type(8)))  float  v8f;

// Pointer types for the async global->LDS copy builtin:
//   void __builtin_amdgcn_global_load_async_to_lds_b128(v4i AS1*, v4i AS3*, Imm, Imm)
typedef int v4i_ __attribute__((vector_size(16)));
typedef __attribute__((address_space(1))) v4i_* gptr_v4i;
typedef __attribute__((address_space(3))) v4i_* lptr_v4i;

#define B_DIM 128
#define N_DIM 256
#define D_DIM 512
#define H_DIM 1024
#define L_DIM 3

#if defined(__has_builtin)
#  if __has_builtin(__builtin_amdgcn_global_load_async_to_lds_b128)
#    define USE_ASYNC_COPY 1
#  endif
#  if __has_builtin(__builtin_amdgcn_s_wait_asynccnt)
#    define HAVE_WAIT_ASYNC 1
#  endif
#endif
#ifndef USE_ASYNC_COPY
#  define USE_ASYNC_COPY 0
#endif

static __device__ __forceinline__ void wait_async_copies() {
#if USE_ASYNC_COPY
#  if defined(HAVE_WAIT_ASYNC)
    __builtin_amdgcn_s_wait_asynccnt(0);
#  else
    asm volatile("s_wait_asynccnt 0x0" ::: "memory");
#  endif
#endif
}

static __device__ __forceinline__ unsigned short f32_to_bf16(float f) {
    unsigned int u = __float_as_uint(f);
    unsigned int r = u + 0x7FFFu + ((u >> 16) & 1u);   // round-to-nearest-even
    return (unsigned short)(r >> 16);
}

// ---------------------------------------------------------------------------
// Row softmax of adjacency [256x256]: fp32 result to d_out, bf16 copy to ws.
// ---------------------------------------------------------------------------
__global__ __launch_bounds__(256) void softmax_adj(const float* __restrict__ adj,
                                                   float* __restrict__ out_f32,
                                                   unsigned short* __restrict__ out_bf) {
    __shared__ float red[256];
    const int row = blockIdx.x, t = threadIdx.x;
    const float v = adj[row * N_DIM + t];
    red[t] = v; __syncthreads();
    for (int s = 128; s > 0; s >>= 1) {
        if (t < s) red[t] = fmaxf(red[t], red[t + s]);
        __syncthreads();
    }
    const float mx = red[0]; __syncthreads();
    const float e = __expf(v - mx);
    red[t] = e; __syncthreads();
    for (int s = 128; s > 0; s >>= 1) {
        if (t < s) red[t] += red[t + s];
        __syncthreads();
    }
    const float r = e * (1.0f / red[0]);
    out_f32[row * N_DIM + t] = r;
    out_bf[row * N_DIM + t] = f32_to_bf16(r);
}

__global__ __launch_bounds__(256) void convert_x(const float* __restrict__ x,
                                                 float* __restrict__ x_out,
                                                 unsigned short* __restrict__ x_bf, int n) {
    const int i = blockIdx.x * 256 + threadIdx.x;
    if (i < n) { const float v = x[i]; x_out[i] = v; x_bf[i] = f32_to_bf16(v); }
}

__global__ __launch_bounds__(256) void f2bf(const float* __restrict__ in,
                                            unsigned short* __restrict__ out, int n) {
    const int i = blockIdx.x * 256 + threadIdx.x;
    if (i < n) out[i] = f32_to_bf16(in[i]);
}

__global__ __launch_bounds__(256) void pool_mean(const float* __restrict__ x,
                                                 float* __restrict__ pooled) {
    const int i = blockIdx.x * 256 + threadIdx.x;   // b*512 + d
    const int b = i >> 9, d = i & 511;
    const float* p = x + (long long)b * (N_DIM * D_DIM) + d;
    float s = 0.f;
    #pragma unroll 8
    for (int n = 0; n < N_DIM; ++n) s += p[n * D_DIM];
    pooled[i] = s * (1.0f / N_DIM);
}

// ---------------------------------------------------------------------------
// Tiled WMMA GEMM: C[M,N] = act(A[M,K] * B + bias)
//   A row-major bf16.  BT=true : B stored [N][K] (weights, y = x*W^T)
//                      BT=false: B stored [K][N] (x tiles for aggregation)
//   ACT=0: store bf16 C
//   ACT=1: bias + ReLU, store bf16 C
//   ACT=2: bias + residual: Xio += val (fp32), Xbf = bf16(Xio)   (C unused)
//
// Block tile 128(M) x 128(N), K-step 32, double-buffered LDS (2x8KB A, 2x8KB B).
// 8 waves (4x2), each wave 32x64 via 2x4 v_wmma_f32_16x16x32_bf16.
// Global->LDS staging via GLOBAL_LOAD_ASYNC_TO_LDS_B128 (ASYNCcnt) when the
// builtin exists; otherwise register-staged loads + ds_store.
// M%128==0, N%128==0, K%32==0 (always true here).
// ---------------------------------------------------------------------------
template <int ACT, bool BT>
__global__ __launch_bounds__(256) void gemm_wmma(
    const unsigned short* __restrict__ A, long long sA,
    const unsigned short* __restrict__ B, long long sB,
    const float* __restrict__ bias,
    unsigned short* __restrict__ C, long long sC,
    float* __restrict__ Xio, unsigned short* __restrict__ Xbf,
    int M, int N, int K) {

    __shared__ __align__(16) unsigned short As[2][128 * 32];
    __shared__ __align__(16) unsigned short Bs[2][128 * 32];   // stored [n][k]

    const int t = threadIdx.x;
    const int lane = t & 31;
    const int wave = t >> 5;
    const int hi = (lane >> 4) & 1;      // lanes 16..31
    const int l15 = lane & 15;
    const int wm = wave >> 1;            // 0..3  -> rows wm*32..+32
    const int wn = wave & 1;             // 0..1  -> cols wn*64..+64
    const int m0 = blockIdx.y * 128;
    const int n0 = blockIdx.x * 128;
    const long long batch = blockIdx.z;

    const unsigned short* Ab = A + batch * sA;
    const unsigned short* Bb = B + batch * sB;

    // Tile-copy indices: 8KB tile = 512 x b128; 2 chunks per thread.
    const int ar0 = t >> 2,         ap0 = (t & 3) * 8;
    const int ar1 = (t + 256) >> 2, ap1 = ((t + 256) & 3) * 8;
    // BT=false transpose-gather indices: 16 u16 per thread.
    const int gn = t >> 1, gh = (t & 1) * 16;

    uint4 stA0, stA1, stB0, stB1;        // sync-path staging
    unsigned short stG[16];              // BT=false staging

    auto issue_loads = [&](int nbuf, int kk) {
#if USE_ASYNC_COPY
        __builtin_amdgcn_global_load_async_to_lds_b128(
            (gptr_v4i)(Ab + (long long)(m0 + ar0) * K + kk + ap0),
            (lptr_v4i)(&As[nbuf][ar0 * 32 + ap0]), 0, 0);
        __builtin_amdgcn_global_load_async_to_lds_b128(
            (gptr_v4i)(Ab + (long long)(m0 + ar1) * K + kk + ap1),
            (lptr_v4i)(&As[nbuf][ar1 * 32 + ap1]), 0, 0);
        if (BT) {
            __builtin_amdgcn_global_load_async_to_lds_b128(
                (gptr_v4i)(Bb + (long long)(n0 + ar0) * K + kk + ap0),
                (lptr_v4i)(&Bs[nbuf][ar0 * 32 + ap0]), 0, 0);
            __builtin_amdgcn_global_load_async_to_lds_b128(
                (gptr_v4i)(Bb + (long long)(n0 + ar1) * K + kk + ap1),
                (lptr_v4i)(&Bs[nbuf][ar1 * 32 + ap1]), 0, 0);
        }
#else
        (void)nbuf;
        stA0 = *reinterpret_cast<const uint4*>(Ab + (long long)(m0 + ar0) * K + kk + ap0);
        stA1 = *reinterpret_cast<const uint4*>(Ab + (long long)(m0 + ar1) * K + kk + ap1);
        if (BT) {
            stB0 = *reinterpret_cast<const uint4*>(Bb + (long long)(n0 + ar0) * K + kk + ap0);
            stB1 = *reinterpret_cast<const uint4*>(Bb + (long long)(n0 + ar1) * K + kk + ap1);
        }
#endif
        if (!BT) {
            #pragma unroll
            for (int j = 0; j < 16; ++j)
                stG[j] = Bb[(long long)(kk + gh + j) * N + (n0 + gn)];
        }
    };

    auto finish_loads = [&](int nbuf) {
#if !USE_ASYNC_COPY
        *reinterpret_cast<uint4*>(&As[nbuf][ar0 * 32 + ap0]) = stA0;
        *reinterpret_cast<uint4*>(&As[nbuf][ar1 * 32 + ap1]) = stA1;
        if (BT) {
            *reinterpret_cast<uint4*>(&Bs[nbuf][ar0 * 32 + ap0]) = stB0;
            *reinterpret_cast<uint4*>(&Bs[nbuf][ar1 * 32 + ap1]) = stB1;
        }
#endif
        if (!BT) {
            #pragma unroll
            for (int j = 0; j < 16; ++j)
                Bs[nbuf][gn * 32 + gh + j] = stG[j];
        }
    };

    union Frag { unsigned int u[8]; v16bf v; };
    v8f acc[2][4] = {};

    // Prologue: fill buffer 0.
    issue_loads(0, 0);
    finish_loads(0);
    wait_async_copies();
    __syncthreads();

    int buf = 0;
    for (int k0 = 0; k0 < K; k0 += 32) {
        const bool nxt = (k0 + 32) < K;
        if (nxt) issue_loads(buf ^ 1, k0 + 32);

        // ---- compute on LDS buffer `buf` ----
        {
            Frag af[2], bfr[4];
            #pragma unroll
            for (int mt = 0; mt < 2; ++mt) {
                const int m = wm * 32 + mt * 16 + l15;
                #pragma unroll
                for (int v = 0; v < 8; ++v) {
                    // ISA 16-bit A 16x32 layout
                    const int k = ((v & 4) << 2) + hi * 8 + 2 * (v & 3);
                    af[mt].u[v] = *reinterpret_cast<const unsigned int*>(&As[buf][m * 32 + k]);
                }
            }
            #pragma unroll
            for (int nt = 0; nt < 4; ++nt) {
                const int n = wn * 64 + nt * 16 + l15;
                #pragma unroll
                for (int v = 0; v < 8; ++v) {
                    // ISA 16-bit B 32x16 layout
                    const int k = ((v & 4) << 2) + hi * 8 + 2 * (v & 3);
                    bfr[nt].u[v] = *reinterpret_cast<const unsigned int*>(&Bs[buf][n * 32 + k]);
                }
            }
            #pragma unroll
            for (int mt = 0; mt < 2; ++mt)
                #pragma unroll
                for (int nt = 0; nt < 4; ++nt)
                    acc[mt][nt] = __builtin_amdgcn_wmma_f32_16x16x32_bf16(
                        false, af[mt].v, false, bfr[nt].v,
                        (short)0, acc[mt][nt], false, false);
        }

        if (nxt) {
            finish_loads(buf ^ 1);
            wait_async_copies();
            __syncthreads();
        }
        buf ^= 1;
    }

    // Epilogue. C/D layout: VGPR r -> row r (lanes 0-15) / row 8+r (lanes 16-31).
    unsigned short* Cb = C + batch * sC;
    #pragma unroll
    for (int mt = 0; mt < 2; ++mt) {
        #pragma unroll
        for (int nt = 0; nt < 4; ++nt) {
            const int colg = n0 + wn * 64 + nt * 16 + l15;
            #pragma unroll
            for (int r = 0; r < 8; ++r) {
                const int rowg = m0 + wm * 32 + mt * 16 + r + hi * 8;
                float v = acc[mt][nt][r];
                const long long idx = (long long)rowg * N + colg;
                if (ACT == 0) {
                    Cb[idx] = f32_to_bf16(v);
                } else if (ACT == 1) {
                    v += bias[colg];
                    v = v > 0.f ? v : 0.f;
                    Cb[idx] = f32_to_bf16(v);
                } else {
                    v += bias[colg];
                    const float xn = Xio[idx] + v;
                    Xio[idx] = xn;
                    Xbf[idx] = f32_to_bf16(xn);
                }
            }
        }
    }
}

// ---------------------------------------------------------------------------
extern "C" void kernel_launch(void* const* d_in, const int* in_sizes, int n_in,
                              void* d_out, int out_size, void* d_ws, size_t ws_size,
                              hipStream_t stream) {
    const float* x_in = (const float*)d_in[0];
    const float* adj  = (const float*)d_in[1];
    const float* W1   = (const float*)d_in[2];
    const float* b1   = (const float*)d_in[3];
    const float* W2   = (const float*)d_in[4];
    const float* b2   = (const float*)d_in[5];

    const int XN = B_DIM * N_DIM * D_DIM;          // 16,777,216
    float* out_x    = (float*)d_out;
    float* out_pool = out_x + XN;                  // [B, D]
    float* out_adj  = out_pool + B_DIM * D_DIM;    // [N, N]

    char* ws = (char*)d_ws;
    unsigned short* x_bf  = (unsigned short*)(ws);                       // 32 MB
    unsigned short* nb_bf = (unsigned short*)(ws + 33554432ll);          // 32 MB
    unsigned short* h_bf  = (unsigned short*)(ws + 67108864ll);          // 64 MB
    unsigned short* adj_bf= (unsigned short*)(ws + 134217728ll);         // 128 KB
    unsigned short* w1_bf = (unsigned short*)(ws + 134217728ll + 131072ll);
    unsigned short* w2_bf = (unsigned short*)(ws + 134217728ll + 131072ll + 3145728ll);

    softmax_adj<<<N_DIM, 256, 0, stream>>>(adj, out_adj, adj_bf);
    convert_x<<<XN / 256, 256, 0, stream>>>(x_in, out_x, x_bf, XN);
    f2bf<<<(L_DIM * H_DIM * D_DIM) / 256, 256, 0, stream>>>(W1, w1_bf, L_DIM * H_DIM * D_DIM);
    f2bf<<<(L_DIM * D_DIM * H_DIM) / 256, 256, 0, stream>>>(W2, w2_bf, L_DIM * D_DIM * H_DIM);

    const long long xStride = (long long)N_DIM * D_DIM;   // 131072 elems
    for (int l = 0; l < L_DIM; ++l) {
        // 1) neighbor[b] = adj_norm @ x[b]   (M=256, N=512, K=256, batch=128)
        {
            dim3 g(D_DIM / 128, N_DIM / 128, B_DIM);
            gemm_wmma<0, false><<<g, 256, 0, stream>>>(
                adj_bf, 0ll, x_bf, xStride, nullptr,
                nb_bf, xStride, nullptr, nullptr,
                N_DIM, D_DIM, N_DIM);
        }
        // 2) h = relu(neighbor @ W1^T + b1)  (M=32768, N=1024, K=512)
        {
            dim3 g(H_DIM / 128, (B_DIM * N_DIM) / 128, 1);
            gemm_wmma<1, true><<<g, 256, 0, stream>>>(
                nb_bf, 0ll, w1_bf + (long long)l * H_DIM * D_DIM, 0ll, b1 + l * H_DIM,
                h_bf, 0ll, nullptr, nullptr,
                B_DIM * N_DIM, H_DIM, D_DIM);
        }
        // 3) x += h @ W2^T + b2              (M=32768, N=512, K=1024)
        {
            dim3 g(D_DIM / 128, (B_DIM * N_DIM) / 128, 1);
            gemm_wmma<2, true><<<g, 256, 0, stream>>>(
                h_bf, 0ll, w2_bf + (long long)l * D_DIM * H_DIM, 0ll, b2 + l * D_DIM,
                nb_bf /*unused*/, 0ll, out_x, x_bf,
                B_DIM * N_DIM, D_DIM, H_DIM);
        }
    }

    pool_mean<<<(B_DIM * D_DIM) / 256, 256, 0, stream>>>(out_x, out_pool);
}
